// DynamicGraphConstructor_57818849739045
// MI455X (gfx1250) — matrix operand order repaired and checked by reading, not compiled, not run on hardware
//
#include <hip/hip_runtime.h>
#include <math.h>

#define B_ 2
#define L_ 12
#define N_ 2048
#define C_ 3
#define H_ 64
#define ND_ 32
#define TE_ 10
#define F_ 116           // H + ND + 2*TE
#define H2_ 128          // 2*H
#define BN_EPS 1e-5f
#define SCALE 0.125f     // 1/sqrt(H)
#define OUT1 ((size_t)B_ * N_ * 3 * N_)   // elements per output tensor

typedef __attribute__((ext_vector_type(16))) _Float16 v16h;
typedef __attribute__((ext_vector_type(8)))  _Float16 h8;
typedef __attribute__((ext_vector_type(8)))  float    v8f;

// Assemble a WMMA 16-element f16 fragment from two contiguous 16B chunks.
__device__ __forceinline__ v16h load_frag(const _Float16* p0, const _Float16* p1) {
  h8 lo = *(const h8*)p0;
  h8 hi = *(const h8*)p1;
  v16h r;
#pragma unroll
  for (int i = 0; i < 8; ++i) { r[i] = lo[i]; r[i + 8] = hi[i]; }
  return r;
}

__device__ __forceinline__ v8f wmma_f16(v16h a, v16h b, v8f c) {
  // (neg_a, A, neg_b, B, c_mod, C, reuse_a, reuse_b)
  return __builtin_amdgcn_wmma_f32_16x16x32_f16(false, a, false, b, (short)0, c,
                                                false, false);
}

// ---------------------------------------------------------------------------
// Kernel 1: fused node MLP (W1+BN+ReLU+W2) and Q/K projections for both
// graphs. One block per (b,n), 128 threads. Writes Q/K as f16 [g][b][n][H].
// ---------------------------------------------------------------------------
__global__ void feat_qk_kernel(const float* __restrict__ hist,
                               const float* __restrict__ embd,
                               const float* __restrict__ embu,
                               const float* __restrict__ tidf,
                               const float* __restrict__ diwf,
                               const float* __restrict__ W1,
                               const float* __restrict__ b1,
                               const float* __restrict__ gam,
                               const float* __restrict__ bet,
                               const float* __restrict__ mean,
                               const float* __restrict__ var,
                               const float* __restrict__ W2,
                               const float* __restrict__ b2,
                               const float* __restrict__ WQ,
                               const float* __restrict__ WK,
                               _Float16* __restrict__ Qws,
                               _Float16* __restrict__ Kws) {
  __shared__ float sx[L_];
  __shared__ float st[2 * TE_];
  __shared__ float sh[H2_];
  __shared__ float sd[H_];
  int blk = blockIdx.x;
  int b = blk >> 11;           // N_ = 2048
  int n = blk & (N_ - 1);
  int t = threadIdx.x;

  if (t < L_) sx[t] = hist[(((size_t)b * L_ + t) * N_ + n) * C_ + 0];
  if (t >= 16 && t < 16 + TE_)
    st[t - 16] = tidf[(((size_t)b * L_ + (L_ - 1)) * N_ + n) * TE_ + (t - 16)];
  if (t >= 32 && t < 32 + TE_)
    st[TE_ + (t - 32)] = diwf[(((size_t)b * L_ + (L_ - 1)) * N_ + n) * TE_ + (t - 32)];
  __syncthreads();

  // h = BN(ReLU(W1 @ x + b1))
  float a = b1[t];
#pragma unroll
  for (int l = 0; l < L_; ++l) a += W1[t * L_ + l] * sx[l];
  a = fmaxf(a, 0.0f);
  a = (a - mean[t]) * rsqrtf(var[t] + BN_EPS) * gam[t] + bet[t];
  sh[t] = a;
  __syncthreads();

  if (t < H_) {
    float d = b2[t];
#pragma unroll 8
    for (int k = 0; k < H2_; ++k) d += W2[t * H2_ + k] * sh[k];
    sd[t] = d;
  }
  __syncthreads();

  int h = t & (H_ - 1);
  int which = t >> 6;                         // 0 -> Q, 1 -> K
  const float* W = which ? WK : WQ;
  float base = 0.0f;
#pragma unroll 8
  for (int f = 0; f < H_; ++f) base += W[h * F_ + f] * sd[f];
#pragma unroll
  for (int j = 0; j < TE_; ++j) base += W[h * F_ + H_ + j] * st[j];
#pragma unroll
  for (int j = 0; j < TE_; ++j) base += W[h * F_ + H_ + TE_ + j] * st[TE_ + j];
  float v1 = base, v2 = base;
#pragma unroll 8
  for (int j = 0; j < ND_; ++j) {
    float w = W[h * F_ + H_ + 2 * TE_ + j];
    v1 += w * embd[(size_t)n * ND_ + j];
    v2 += w * embu[(size_t)n * ND_ + j];
  }
  _Float16* dst = which ? Kws : Qws;
  dst[((size_t)(0 * B_ + b) * N_ + n) * H_ + h] = (_Float16)v1;  // graph 0
  dst[((size_t)(1 * B_ + b) * N_ + n) * H_ + h] = (_Float16)v2;  // graph 1
}

// ---------------------------------------------------------------------------
// Kernels 2-4: attention passes, MODE as template param so each instantiation
// has a branch-free streaming loop. Each wave owns 16 query rows and streams
// across all 2048 key columns, recomputing S = Q K^T with WMMA each pass.
//   MODE 1: row max of scaled scores          -> mvec
//   MODE 2: row sum of (mask+eps)*exp(s-m)    -> rvec
//   MODE 3: A = t / r ; f16 ws copy (with diag, regular store -> stays in L2
//           for the A@A GEMM) + localized f32 out (diag zeroed, 3x, NT store)
// Softmax denominator cancels against the renormalization, so m,r suffice.
// ---------------------------------------------------------------------------
template <int MODE>
__global__ void attn_pass_kernel(const _Float16* __restrict__ Qws,
                                 const _Float16* __restrict__ Kws,
                                 const float* __restrict__ mask0,
                                 const float* __restrict__ mask1,
                                 float* __restrict__ mvec,
                                 float* __restrict__ rvec,
                                 _Float16* __restrict__ Abf,
                                 float* __restrict__ out) {
  int lane = threadIdx.x & 31;
  int wid = threadIdx.x >> 5;
  int z = blockIdx.y;
  int b = z & 1, g = z >> 1;
  int gb = g * B_ + b;
  int row0 = blockIdx.x * 64 + wid * 16;
  int lh = lane & 15, hi = lane >> 4;

  const _Float16* Q = Qws + (size_t)gb * N_ * H_;
  const _Float16* Km = Kws + (size_t)gb * N_ * H_;
  const float* mask = g ? mask1 : mask0;

  // Preload the two A-fragments (k = 0..31, 32..63) for our 16 Q rows.
  const _Float16* qrow = Q + (size_t)(row0 + lh) * H_;
  v16h aq0 = load_frag(qrow + 0 + hi * 8, qrow + 16 + hi * 8);
  v16h aq1 = load_frag(qrow + 32 + hi * 8, qrow + 48 + hi * 8);

  float acc8[8];
#pragma unroll
  for (int r = 0; r < 8; ++r) acc8[r] = (MODE == 1) ? -3.0e38f : 0.0f;

  float mreg[8], invr[8];
  if (MODE >= 2) {
#pragma unroll
    for (int r = 0; r < 8; ++r)
      mreg[r] = mvec[(size_t)gb * N_ + row0 + r + 8 * hi];
  }
  if (MODE == 3) {
#pragma unroll
    for (int r = 0; r < 8; ++r) {
      float rs = rvec[(size_t)gb * N_ + row0 + r + 8 * hi];
      float iv = 1.0f / rs;
      if (!(fabsf(iv) <= 3.0e38f)) iv = 0.0f;   // mirror isfinite guard
      invr[r] = iv;
    }
  }

  for (int nt = 0; nt < N_ / 16; ++nt) {
    int n0 = nt * 16;
    // B-fragment = contiguous rows of K (columns of K^T): no transpose needed.
    const _Float16* krow = Km + (size_t)(n0 + lh) * H_;
    v16h bk0 = load_frag(krow + 0 + hi * 8, krow + 16 + hi * 8);
    v16h bk1 = load_frag(krow + 32 + hi * 8, krow + 48 + hi * 8);
    v8f c = {};
    c = wmma_f16(aq0, bk0, c);
    c = wmma_f16(aq1, bk1, c);

    if (MODE == 1) {
#pragma unroll
      for (int r = 0; r < 8; ++r) acc8[r] = fmaxf(acc8[r], c[r] * SCALE);
    } else {
      int col = n0 + lh;
#pragma unroll
      for (int r = 0; r < 8; ++r) {
        int row = row0 + r + 8 * hi;
        float s = c[r] * SCALE;
        float e = __expf(s - mreg[r]) * (mask[(size_t)row * N_ + col] + 1e-7f);
        if (MODE == 2) {
          acc8[r] += e;
        } else {
          float av = e * invr[r];
          Abf[(size_t)gb * N_ * N_ + (size_t)row * N_ + col] = (_Float16)av;
          float ov = (row == col) ? 0.0f : av;       // offdiag for output 2g
          size_t ob = (size_t)(2 * g) * OUT1 +
                      ((size_t)b * N_ + row) * (size_t)(3 * N_) + col;
          __builtin_nontemporal_store(ov, out + ob);
          __builtin_nontemporal_store(ov, out + ob + N_);
          __builtin_nontemporal_store(ov, out + ob + 2 * N_);
        }
      }
    }
  }

  if (MODE <= 2) {
#pragma unroll
    for (int r = 0; r < 8; ++r) {
      float v = acc8[r];
#pragma unroll
      for (int m = 1; m <= 8; m <<= 1) {
        float o = __shfl_xor(v, m, 32);
        v = (MODE == 1) ? fmaxf(v, o) : (v + o);
      }
      acc8[r] = v;
    }
    if (lh == 0) {  // lanes 0 and 16 write rows 0..7 / 8..15
      float* dst = (MODE == 1) ? mvec : rvec;
#pragma unroll
      for (int r = 0; r < 8; ++r)
        dst[(size_t)gb * N_ + row0 + r + 8 * hi] = acc8[r];
    }
  }
}

// ---------------------------------------------------------------------------
// Kernel 5: C = A @ A (f16 in, f32 accumulate) per (b,g); 2048^3 each.
// 128 threads = 4 waves; block tile 64x64 (wave tile 32x32 = 4 accumulators).
// A-tile staged row-major in LDS; B-tile staged TRANSPOSED so B-fragments are
// contiguous ds_load_b128 pairs. Epilogue: zero diag, localized 3x NT write.
// ---------------------------------------------------------------------------
#define LDA 48  // padded LDS row stride in halves (96B, multiple of 16B)

__global__ void __launch_bounds__(128)
gemm_a2_kernel(const _Float16* __restrict__ Abf, float* __restrict__ out) {
  __shared__ _Float16 As[64 * LDA];
  __shared__ _Float16 Bt[64 * LDA];

  int t = threadIdx.x;
  int lane = t & 31, wid = t >> 5;
  int lh = lane & 15, hi = lane >> 4;
  int z = blockIdx.z;
  int b = z & 1, g = z >> 1;
  int tileM = blockIdx.y * 64;
  int tileN = blockIdx.x * 64;
  int waveM = (wid >> 1) * 32;
  int waveN = (wid & 1) * 32;

  const _Float16* A = Abf + (size_t)(g * B_ + b) * N_ * N_;

  v8f zero = {};
  v8f acc[2][2];
  acc[0][0] = zero; acc[0][1] = zero; acc[1][0] = zero; acc[1][1] = zero;

  int ar = t >> 1, ach = (t & 1) * 16;   // A-tile staging: row, 16-half chunk
  int kp = t >> 3, n8 = (t & 7) * 8;     // B-tile staging: k-pair, 8 cols

  for (int kt = 0; kt < N_ / 32; ++kt) {
    int k0 = kt * 32;

    // Stage A tile (64x32, row-major)
    {
      const h8* src = (const h8*)(A + (size_t)(tileM + ar) * N_ + k0 + ach);
      h8* dst = (h8*)(As + ar * LDA + ach);
      dst[0] = src[0];
      dst[1] = src[1];
    }
    // Stage B tile transposed: Bt[n][k] <- A[k0+k][tileN+n], k-pairs packed b32
    {
      const _Float16* s0 = A + (size_t)(k0 + 2 * kp) * N_ + tileN + n8;
      h8 r0 = *(const h8*)s0;
      h8 r1 = *(const h8*)(s0 + N_);
#pragma unroll
      for (int j = 0; j < 8; ++j) {
        union { _Float16 h[2]; unsigned u; } pk;
        pk.h[0] = r0[j];
        pk.h[1] = r1[j];
        *(unsigned*)&Bt[(n8 + j) * LDA + 2 * kp] = pk.u;
      }
    }
    if (kt + 1 < N_ / 32) {  // prefetch next k-tile (global_prefetch_b8)
      __builtin_prefetch(A + (size_t)(tileM + ar) * N_ + k0 + 32 + ach, 0, 3);
      __builtin_prefetch(A + (size_t)(k0 + 32 + 2 * kp) * N_ + tileN + n8, 0, 3);
    }
    __syncthreads();

    v16h af[2], bf[2];
#pragma unroll
    for (int i = 0; i < 2; ++i) {
      const _Float16* p = As + (waveM + i * 16 + lh) * LDA + hi * 8;
      af[i] = load_frag(p, p + 16);
    }
#pragma unroll
    for (int j = 0; j < 2; ++j) {
      const _Float16* p = Bt + (waveN + j * 16 + lh) * LDA + hi * 8;
      bf[j] = load_frag(p, p + 16);
    }
#pragma unroll
    for (int i = 0; i < 2; ++i)
#pragma unroll
      for (int j = 0; j < 2; ++j) acc[i][j] = wmma_f16(af[i], bf[j], acc[i][j]);

    __syncthreads();
  }

  // Epilogue: C layout elem r -> (m = r + 8*hi, n = lh); localized 3x write.
#pragma unroll
  for (int i = 0; i < 2; ++i)
#pragma unroll
    for (int j = 0; j < 2; ++j)
#pragma unroll
      for (int r = 0; r < 8; ++r) {
        int m = tileM + waveM + i * 16 + r + 8 * hi;
        int n = tileN + waveN + j * 16 + lh;
        float v = (m == n) ? 0.0f : acc[i][j][r];
        size_t ob = (size_t)(2 * g + 1) * OUT1 +
                    ((size_t)b * N_ + m) * (size_t)(3 * N_) + n;
        __builtin_nontemporal_store(v, out + ob);
        __builtin_nontemporal_store(v, out + ob + N_);
        __builtin_nontemporal_store(v, out + ob + 2 * N_);
      }
}

// ---------------------------------------------------------------------------
extern "C" void kernel_launch(void* const* d_in, const int* in_sizes, int n_in,
                              void* d_out, int out_size, void* d_ws,
                              size_t ws_size, hipStream_t stream) {
  const float* hist  = (const float*)d_in[0];
  const float* embd  = (const float*)d_in[1];
  const float* embu  = (const float*)d_in[2];
  const float* tidf  = (const float*)d_in[3];
  const float* diwf  = (const float*)d_in[4];
  const float* mask0 = (const float*)d_in[5];
  const float* mask1 = (const float*)d_in[6];
  const float* W1    = (const float*)d_in[7];
  const float* b1    = (const float*)d_in[8];
  const float* gam   = (const float*)d_in[9];
  const float* bet   = (const float*)d_in[10];
  const float* mean  = (const float*)d_in[11];
  const float* var   = (const float*)d_in[12];
  const float* W2    = (const float*)d_in[13];
  const float* b2    = (const float*)d_in[14];
  const float* WQ    = (const float*)d_in[15];
  const float* WK    = (const float*)d_in[16];
  float* out = (float*)d_out;

  // Workspace layout (~35.6 MB): Q/K f16, row stats f32, A f16.
  const size_t nQK = (size_t)2 * B_ * N_ * H_;     // per Q or K, halves
  _Float16* Qws = (_Float16*)d_ws;
  _Float16* Kws = Qws + nQK;
  float* mvec = (float*)(Kws + nQK);
  float* rvec = mvec + (size_t)2 * B_ * N_;
  _Float16* Abf = (_Float16*)(rvec + (size_t)2 * B_ * N_);

  // 1) fused MLP + Q/K projections
  feat_qk_kernel<<<B_ * N_, 128, 0, stream>>>(hist, embd, embu, tidf, diwf, W1,
                                              b1, gam, bet, mean, var, W2, b2,
                                              WQ, WK, Qws, Kws);
  // 2-4) attention: rowmax, masked-exp rowsum, normalized write
  dim3 agrid(N_ / 64, 2 * B_);
  attn_pass_kernel<1><<<agrid, 128, 0, stream>>>(Qws, Kws, mask0, mask1, mvec,
                                                 rvec, Abf, out);
  attn_pass_kernel<2><<<agrid, 128, 0, stream>>>(Qws, Kws, mask0, mask1, mvec,
                                                 rvec, Abf, out);
  attn_pass_kernel<3><<<agrid, 128, 0, stream>>>(Qws, Kws, mask0, mask1, mvec,
                                                 rvec, Abf, out);
  // 5) A @ A with localized epilogue
  dim3 ggrid(N_ / 64, N_ / 64, 2 * B_);
  gemm_a2_kernel<<<ggrid, 128, 0, stream>>>(Abf, out);
}
